// AblationMamba_29996051595452
// MI455X (gfx1250) — matrix operand order repaired and checked by reading, not compiled, not run on hardware
//
#include <hip/hip_runtime.h>
#include <hip/hip_bf16.h>

// ---------------------------------------------------------------------------
// Model constants
// ---------------------------------------------------------------------------
#define BB       4
#define TT       512
#define TOK      (BB * TT)          // 2048
#define NCH      64
#define DM       769                // D_MODEL
#define DM_P     800                // padded to 32 (25 * 32)
#define DI       1538               // D_INNER
#define DI_P     1568               // padded to 32 (49 * 32)
#define DI_NP    1552               // padded to 16 (97 * 16)
#define NXZ      3076               // 2 * D_INNER
#define NXZ_P    3088               // padded to 16 (193 * 16)
#define DTR      49
#define DTR_P    64                 // padded to 32
#define DST      64
#define NDBC     177                // DTR + 2*DST
#define NDBC_P   192                // padded to 16
#define DM_NP    784                // DM padded to 16 (49 * 16)
#define DCONV    4
#define LLAYERS  2

typedef __attribute__((ext_vector_type(16))) _Float16 v16h;
typedef __attribute__((ext_vector_type(8)))  _Float16 v8h;
typedef __attribute__((ext_vector_type(8)))  float    v8f;

union F16x16 { v16h v; v8h h[2]; };

__device__ __forceinline__ float silu_f(float x) {
    return x * (1.0f / (1.0f + __expf(-x)));
}
__device__ __forceinline__ float softplus_f(float x) {
    return (x > 20.0f) ? x : log1pf(__expf(x));
}

// ---------------------------------------------------------------------------
// Front-end: pointwise conv (64 -> 128) + SiLU.  h1 layout (B,128,T)
// ---------------------------------------------------------------------------
__global__ void k_pointconv(const float* __restrict__ x,   // (B,T,65)
                            const float* __restrict__ pw,  // (128,64,1)
                            const float* __restrict__ pb,  // (128)
                            float* __restrict__ h1) {      // (B,128,T)
    int idx = blockIdx.x * blockDim.x + threadIdx.x;       // B*128*T
    if (idx >= BB * 128 * TT) return;
    int t = idx & (TT - 1);
    int c = (idx / TT) & 127;
    int b = idx / (128 * TT);
    const float* xr = x + ((size_t)(b * TT + t)) * (NCH + 1);
    const float* wr = pw + c * NCH;
    float s = pb[c];
    #pragma unroll 8
    for (int i = 0; i < NCH; ++i) s += xr[i] * wr[i];
    h1[idx] = silu_f(s);
}

// ---------------------------------------------------------------------------
// Front-end: 1D conv (128 -> 256, kernel kk, SAME) + SiLU, scattered into
// h (B,T,769) at channel offset 'coff'
// ---------------------------------------------------------------------------
__global__ void k_conv1d(const float* __restrict__ h1,   // (B,128,T)
                         const float* __restrict__ w,    // (256,128,kk)
                         const float* __restrict__ bias, // (256)
                         int kk, int pad, int coff,
                         float* __restrict__ h) {        // (B,T,769)
    int idx = blockIdx.x * blockDim.x + threadIdx.x;     // B*256*T
    if (idx >= BB * 256 * TT) return;
    int t = idx & (TT - 1);
    int c = (idx / TT) & 255;
    int b = idx / (256 * TT);
    float s = bias[c];
    for (int i = 0; i < 128; ++i) {
        const float* src = h1 + ((size_t)(b * 128 + i)) * TT;
        const float* wr  = w + ((size_t)(c * 128 + i)) * kk;
        for (int j = 0; j < kk; ++j) {
            int tt = t + j - pad;
            if (tt >= 0 && tt < TT) s += src[tt] * wr[j];
        }
    }
    h[((size_t)(b * TT + t)) * DM + coff + c] = silu_f(s);
}

// positional channel: h[...,768] = x[...,64]
__global__ void k_copy_pe(const float* __restrict__ x, float* __restrict__ h) {
    int tok = blockIdx.x * blockDim.x + threadIdx.x;
    if (tok >= TOK) return;
    h[(size_t)tok * DM + (DM - 1)] = x[(size_t)tok * (NCH + 1) + NCH];
}

// ---------------------------------------------------------------------------
// RMSNorm over 769, emit f16 row padded to 800
// ---------------------------------------------------------------------------
__global__ void k_rmsnorm_cast(const float* __restrict__ h,
                               const float* __restrict__ w,
                               _Float16* __restrict__ out) {
    __shared__ float red[256];
    int tok = blockIdx.x;
    int tid = threadIdx.x;
    const float* row = h + (size_t)tok * DM;
    float s = 0.0f;
    for (int c = tid; c < DM; c += 256) { float v = row[c]; s += v * v; }
    red[tid] = s;
    __syncthreads();
    for (int st = 128; st > 0; st >>= 1) {
        if (tid < st) red[tid] += red[tid + st];
        __syncthreads();
    }
    float rs = rsqrtf(red[0] / (float)DM + 1e-6f);
    _Float16* orow = out + (size_t)tok * DM_P;
    for (int c = tid; c < DM_P; c += 256)
        orow[c] = (c < DM) ? (_Float16)(row[c] * rs * w[c]) : (_Float16)0.0f;
}

// ---------------------------------------------------------------------------
// Pack weight W (Nreal x Kreal, row-major f32) into wave32 WMMA B fragments:
// frag(ntile,kc): 32 lanes x 16 halfs; lane&15 = column, lane>=16 -> K+16.
// ---------------------------------------------------------------------------
__global__ void k_pack_w(const float* __restrict__ W, _Float16* __restrict__ Bp,
                         int Nreal, int Kreal, int Np, int Kp) {
    int idx = blockIdx.x * blockDim.x + threadIdx.x;
    int total = (Np / 16) * (Kp / 32) * 512;
    if (idx >= total) return;
    int frag   = idx >> 9;
    int within = idx & 511;
    int lane   = within >> 4;
    int e      = within & 15;
    int kchunks = Kp / 32;
    int ntile = frag / kchunks;
    int kc    = frag - ntile * kchunks;
    int n = ntile * 16 + (lane & 15);
    int k = kc * 32 + (lane >> 4) * 16 + e;
    float v = (n < Nreal && k < Kreal) ? W[(size_t)n * Kreal + k] : 0.0f;
    Bp[idx] = (_Float16)v;
}

// ---------------------------------------------------------------------------
// WMMA GEMM:  C(M x Nreal) = A(M x Kp, f16 row-major) * Bp (packed fragments)
// epilogue: optional bias[col], optional softplus (act==1), optional residual.
// One wave per 16x16 output tile; 4 waves per block along M.
// ---------------------------------------------------------------------------
__global__ void k_wmma_gemm(const _Float16* __restrict__ A,
                            const _Float16* __restrict__ Bp,
                            float* __restrict__ C,
                            int M, int Nreal, int Kp,
                            const float* __restrict__ bias,
                            const float* __restrict__ resid,
                            int act) {
    const int lane  = threadIdx.x;           // 0..31
    const int ntile = blockIdx.x;
    const int mtile = blockIdx.y * 4 + threadIdx.y;
    const int kchunks = Kp >> 5;
    const int hi = lane >> 4;                // 0 or 1

    const _Float16* arow  = A + ((size_t)(mtile * 16 + (lane & 15))) * Kp + hi * 16;
    const _Float16* bfrag = Bp + ((size_t)ntile * kchunks) * 512 + lane * 16;

    v8f acc = {0.f, 0.f, 0.f, 0.f, 0.f, 0.f, 0.f, 0.f};
    for (int kc = 0; kc < kchunks; ++kc) {
        F16x16 a, b;
        a.h[0] = *(const v8h*)(arow);
        a.h[1] = *(const v8h*)(arow + 8);
        b.h[0] = *(const v8h*)(bfrag);
        b.h[1] = *(const v8h*)(bfrag + 8);
        __builtin_prefetch(arow + 32, 0, 0);
        __builtin_prefetch(bfrag + 512, 0, 0);
        acc = __builtin_amdgcn_wmma_f32_16x16x32_f16(
            false, a.v, false, b.v, (short)0, acc, false, false);
        arow  += 32;
        bfrag += 512;
    }

    int col = ntile * 16 + (lane & 15);
    if (col >= Nreal) return;
    int rbase = mtile * 16 + hi * 8;
    float bv = bias ? bias[col] : 0.0f;
    #pragma unroll
    for (int r = 0; r < 8; ++r) {
        float v = acc[r] + bv;
        if (act == 1) v = softplus_f(v);
        size_t off = (size_t)(rbase + r) * Nreal + col;
        if (resid) v += resid[off];
        C[off] = v;
    }
}

// ---------------------------------------------------------------------------
// Split xz, depthwise causal conv k=4 + SiLU -> xin (f32 for scan, f16 padded)
// ---------------------------------------------------------------------------
__global__ void k_dwconv_silu(const float* __restrict__ xz,   // (TOK,3076)
                              const float* __restrict__ cw,   // (1538,4)
                              const float* __restrict__ cb,   // (1538)
                              float* __restrict__ xin32,      // (TOK,1538)
                              _Float16* __restrict__ xin16) { // (TOK,1568)
    int idx = blockIdx.x * blockDim.x + threadIdx.x;
    if (idx >= TOK * DI_P) return;
    int tok = idx / DI_P;
    int d   = idx - tok * DI_P;
    if (d >= DI) { xin16[(size_t)tok * DI_P + d] = (_Float16)0.0f; return; }
    int b = tok / TT, t = tok - b * TT;
    float s = cb[d];
    #pragma unroll
    for (int j = 0; j < DCONV; ++j) {
        int tt = t + j - (DCONV - 1);
        if (tt >= 0) s += xz[((size_t)(b * TT + tt)) * NXZ + d] * cw[d * DCONV + j];
    }
    float v = silu_f(s);
    xin32[(size_t)tok * DI + d]   = v;
    xin16[(size_t)tok * DI_P + d] = (_Float16)v;
}

// dbc[:, :49] -> f16 padded to 64 columns
__global__ void k_dt_cast(const float* __restrict__ dbc, _Float16* __restrict__ o) {
    int idx = blockIdx.x * blockDim.x + threadIdx.x;
    if (idx >= TOK * DTR_P) return;
    int tok = idx / DTR_P;
    int k   = idx - tok * DTR_P;
    float v = (k < DTR) ? dbc[(size_t)tok * NDBC + k] : 0.0f;
    o[idx] = (_Float16)v;
}

// ---------------------------------------------------------------------------
// Selective scan: one thread per (b,d); 64-wide state held in registers.
// Fuses u*D skip and SiLU(z) gating; writes f16 row padded to 1568.
// ---------------------------------------------------------------------------
__global__ void k_scan(const float* __restrict__ xin32,   // (TOK,1538)
                       const float* __restrict__ dtb,     // (TOK,1538) softplus'd
                       const float* __restrict__ dbc,     // (TOK,177) B at 49, C at 113
                       const float* __restrict__ xz,      // (TOK,3076) z at 1538
                       const float* __restrict__ A_log,   // (1538,64)
                       const float* __restrict__ Dsk,     // (1538)
                       _Float16* __restrict__ y16) {      // (TOK,1568)
    int d = blockIdx.x * 32 + threadIdx.x;
    int b = blockIdx.y;
    if (d >= DI_P) return;
    if (d >= DI) {
        for (int t = 0; t < TT; ++t)
            y16[((size_t)(b * TT + t)) * DI_P + d] = (_Float16)0.0f;
        return;
    }
    float An[DST], hs[DST];
    #pragma unroll
    for (int n = 0; n < DST; ++n) {
        An[n] = -__expf(A_log[(size_t)d * DST + n]);
        hs[n] = 0.0f;
    }
    const float Dv = Dsk[d];
    for (int t = 0; t < TT; ++t) {
        size_t tok = (size_t)(b * TT + t);
        float u   = xin32[tok * DI + d];
        float dtv = dtb[tok * DI + d];
        const float* Bm = dbc + tok * NDBC + DTR;
        const float* Cm = Bm + DST;
        float du = dtv * u;
        float acc = 0.0f;
        #pragma unroll
        for (int n = 0; n < DST; ++n) {
            float dA = __expf(dtv * An[n]);
            hs[n] = dA * hs[n] + du * Bm[n];
            acc += hs[n] * Cm[n];
        }
        float z = xz[tok * NXZ + DI + d];
        float out = (acc + u * Dv) * silu_f(z);
        y16[tok * DI_P + d] = (_Float16)out;
    }
}

// ---------------------------------------------------------------------------
// Final LayerNorm(769) + 10-class head
// ---------------------------------------------------------------------------
__global__ void k_final(const float* __restrict__ h,
                        const float* __restrict__ lnw, const float* __restrict__ lnb,
                        const float* __restrict__ ow,  const float* __restrict__ ob,
                        float* __restrict__ out) {
    __shared__ float nrm[DM];
    __shared__ float red[256];
    __shared__ float smu, srs;
    int tok = blockIdx.x;
    int tid = threadIdx.x;
    const float* row = h + (size_t)tok * DM;

    float s = 0.0f;
    for (int c = tid; c < DM; c += 256) s += row[c];
    red[tid] = s; __syncthreads();
    for (int st = 128; st > 0; st >>= 1) { if (tid < st) red[tid] += red[tid + st]; __syncthreads(); }
    if (tid == 0) smu = red[0] / (float)DM;
    __syncthreads();
    float mu = smu;

    s = 0.0f;
    for (int c = tid; c < DM; c += 256) { float v = row[c] - mu; s += v * v; }
    red[tid] = s; __syncthreads();
    for (int st = 128; st > 0; st >>= 1) { if (tid < st) red[tid] += red[tid + st]; __syncthreads(); }
    if (tid == 0) srs = rsqrtf(red[0] / (float)DM + 1e-5f);
    __syncthreads();
    float rs = srs;

    for (int c = tid; c < DM; c += 256)
        nrm[c] = (row[c] - mu) * rs * lnw[c] + lnb[c];
    __syncthreads();

    if (tid < 10) {
        const float* wr = ow + (size_t)tid * DM;
        float acc = ob[tid];
        for (int c = 0; c < DM; ++c) acc += nrm[c] * wr[c];
        out[(size_t)tok * 10 + tid] = acc;
    }
}

// ---------------------------------------------------------------------------
// Host orchestration
// ---------------------------------------------------------------------------
extern "C" void kernel_launch(void* const* d_in, const int* in_sizes, int n_in,
                              void* d_out, int out_size, void* d_ws, size_t ws_size,
                              hipStream_t stream) {
    const float* x   = (const float*)d_in[0];
    const float* pw  = (const float*)d_in[1];
    const float* pb  = (const float*)d_in[2];
    const float* c1w = (const float*)d_in[3];
    const float* c1b = (const float*)d_in[4];
    const float* c2w = (const float*)d_in[5];
    const float* c2b = (const float*)d_in[6];
    const float* c3w = (const float*)d_in[7];
    const float* c3b = (const float*)d_in[8];
    const float* lnw = (const float*)d_in[9];
    const float* lnb = (const float*)d_in[10];
    const float* ow  = (const float*)d_in[11];
    const float* obv = (const float*)d_in[12];
    const float* rmsw  = (const float*)d_in[13];  // (L,769)
    const float* inw   = (const float*)d_in[14];  // (L,3076,769)
    const float* dww   = (const float*)d_in[15];  // (L,1538,1,4)
    const float* dwb   = (const float*)d_in[16];  // (L,1538)
    const float* xpw   = (const float*)d_in[17];  // (L,177,1538)
    const float* dtw   = (const float*)d_in[18];  // (L,1538,49)
    const float* dtbias= (const float*)d_in[19];  // (L,1538)
    const float* alog  = (const float*)d_in[20];  // (L,1538,64)
    const float* dsk   = (const float*)d_in[21];  // (L,1538)
    const float* opw   = (const float*)d_in[22];  // (L,769,1538)

    // workspace bump allocator
    char* ws = (char*)d_ws;
    size_t off = 0;
    auto alloc = [&](size_t bytes) -> char* {
        char* p = ws + off;
        off = (off + bytes + 255) & ~(size_t)255;
        return p;
    };
    float*    h1     = (float*)   alloc((size_t)BB * 128 * TT * 4);
    float*    hbuf   = (float*)   alloc((size_t)TOK * DM * 4);
    _Float16* xn16   = (_Float16*)alloc((size_t)TOK * DM_P * 2);
    _Float16* wbuf   = (_Float16*)alloc((size_t)NXZ_P * DM_P * 2);  // max pack
    float*    xz     = (float*)   alloc((size_t)TOK * NXZ * 4);
    float*    xin32  = (float*)   alloc((size_t)TOK * DI * 4);
    _Float16* xin16  = (_Float16*)alloc((size_t)TOK * DI_P * 2);
    float*    dbc    = (float*)   alloc((size_t)TOK * NDBC * 4);
    _Float16* dt16   = (_Float16*)alloc((size_t)TOK * DTR_P * 2);
    float*    dtb    = (float*)   alloc((size_t)TOK * DI * 4);
    _Float16* y16    = (_Float16*)alloc((size_t)TOK * DI_P * 2);

    const dim3 gblk(32, 4);

    // ---- front-end ----
    k_pointconv<<<(BB * 128 * TT + 255) / 256, 256, 0, stream>>>(x, pw, pb, h1);
    int cvblocks = (BB * 256 * TT + 255) / 256;
    k_conv1d<<<cvblocks, 256, 0, stream>>>(h1, c1w, c1b, 3, 1, 0, hbuf);
    k_conv1d<<<cvblocks, 256, 0, stream>>>(h1, c2w, c2b, 9, 4, 256, hbuf);
    k_conv1d<<<cvblocks, 256, 0, stream>>>(h1, c3w, c3b, 27, 13, 512, hbuf);
    k_copy_pe<<<(TOK + 255) / 256, 256, 0, stream>>>(x, hbuf);

    // ---- mamba layers ----
    for (int l = 0; l < LLAYERS; ++l) {
        const float* rw  = rmsw  + (size_t)l * DM;
        const float* iw  = inw   + (size_t)l * NXZ * DM;
        const float* cw  = dww   + (size_t)l * DI * DCONV;
        const float* cb  = dwb   + (size_t)l * DI;
        const float* xw  = xpw   + (size_t)l * NDBC * DI;
        const float* dw  = dtw   + (size_t)l * DI * DTR;
        const float* db  = dtbias+ (size_t)l * DI;
        const float* al  = alog  + (size_t)l * DI * DST;
        const float* ds  = dsk   + (size_t)l * DI;
        const float* opl = opw   + (size_t)l * DM * DI;

        // rmsnorm -> f16
        k_rmsnorm_cast<<<TOK, 256, 0, stream>>>(hbuf, rw, xn16);

        // in_proj: (2048 x 3076) = xn16(2048x800) * W^T
        {
            int total = (NXZ_P / 16) * (DM_P / 32) * 512;
            k_pack_w<<<(total + 255) / 256, 256, 0, stream>>>(iw, wbuf, NXZ, DM, NXZ_P, DM_P);
            k_wmma_gemm<<<dim3(NXZ_P / 16, TOK / 64), gblk, 0, stream>>>(
                xn16, wbuf, xz, TOK, NXZ, DM_P, nullptr, nullptr, 0);
        }

        // depthwise causal conv + silu
        k_dwconv_silu<<<(TOK * DI_P + 255) / 256, 256, 0, stream>>>(xz, cw, cb, xin32, xin16);

        // x_proj: (2048 x 177)
        {
            int total = (NDBC_P / 16) * (DI_P / 32) * 512;
            k_pack_w<<<(total + 255) / 256, 256, 0, stream>>>(xw, wbuf, NDBC, DI, NDBC_P, DI_P);
            k_wmma_gemm<<<dim3(NDBC_P / 16, TOK / 64), gblk, 0, stream>>>(
                xin16, wbuf, dbc, TOK, NDBC, DI_P, nullptr, nullptr, 0);
        }

        // dt projection: (2048 x 1538), bias + softplus epilogue
        {
            k_dt_cast<<<(TOK * DTR_P + 255) / 256, 256, 0, stream>>>(dbc, dt16);
            int total = (DI_NP / 16) * (DTR_P / 32) * 512;
            k_pack_w<<<(total + 255) / 256, 256, 0, stream>>>(dw, wbuf, DI, DTR, DI_NP, DTR_P);
            k_wmma_gemm<<<dim3(DI_NP / 16, TOK / 64), gblk, 0, stream>>>(
                dt16, wbuf, dtb, TOK, DI, DTR_P, db, nullptr, 1);
        }

        // selective scan + skip + gate -> y16
        k_scan<<<dim3(DI_P / 32, BB), 32, 0, stream>>>(xin32, dtb, dbc, xz, al, ds, y16);

        // out_proj: (2048 x 769) + residual (in-place into hbuf)
        {
            int total = (DM_NP / 16) * (DI_P / 32) * 512;
            k_pack_w<<<(total + 255) / 256, 256, 0, stream>>>(opl, wbuf, DM, DI, DM_NP, DI_P);
            k_wmma_gemm<<<dim3(DM_NP / 16, TOK / 64), gblk, 0, stream>>>(
                y16, wbuf, hbuf, TOK, DM, DI_P, nullptr, hbuf, 0);
        }
    }

    // ---- layernorm + head ----
    k_final<<<TOK, 256, 0, stream>>>(hbuf, lnw, lnb, ow, obv, (float*)d_out);
}
